// GraphConvolution_18064632447538
// MI455X (gfx1250) — matrix operand order repaired and checked by reading, not compiled, not run on hardware
//
#include <hip/hip_runtime.h>
#include <hip/hip_fp16.h>

typedef __attribute__((ext_vector_type(16))) _Float16 v16h;
typedef __attribute__((ext_vector_type(8)))  _Float16 v8h;
typedef __attribute__((ext_vector_type(2)))  _Float16 v2h;
typedef __attribute__((ext_vector_type(8)))  float    v8f;

#define BATCH 4
#define NVERT 40000L
#define NEDGE 120000

// ---------------------------------------------------------------------------
// Weight prep: f32 -> f16, WMMA B-fragment order:
// dst[((kt*NT + nt)*32 + lane)*16 + e] = W[kt*32 + (lane>>4)*16 + e][nt*16 + (lane&15)]
// ---------------------------------------------------------------------------
__global__ void swizzle_w_kernel(const float* __restrict__ src, _Float16* __restrict__ dst,
                                 int K, int Nsrc, int NT,
                                 long srcLayerStride, long dstLayerStride) {
    const int layer = blockIdx.y;
    const float* s = src + (long)layer * srcLayerStride;
    _Float16*   d = dst + (long)layer * dstLayerStride;
    const int total = (K >> 5) * NT * 512;
    int idx = blockIdx.x * blockDim.x + threadIdx.x;
    if (idx >= total) return;
    int e    = idx & 15;
    int lane = (idx >> 4) & 31;
    int nt   = (idx >> 9) % NT;
    int kt   = idx / (NT * 512);
    int k = kt * 32 + ((lane >> 4) << 4) + e;
    int n = nt * 16 + (lane & 15);
    float v = (n < Nsrc) ? s[(long)k * Nsrc + n] : 0.0f;
    d[idx] = (_Float16)v;
}

__global__ void bias_kernel(const float* __restrict__ b0, const float* __restrict__ b1,
                            float* __restrict__ dst, int Nsrc, int Npad) {
    const int layer = blockIdx.y;
    int n = blockIdx.x * blockDim.x + threadIdx.x;
    if (n >= Npad) return;
    float v = (n < Nsrc) ? (b0[(long)layer * Nsrc + n] + b1[(long)layer * Nsrc + n]) : 0.0f;
    dst[(long)layer * Npad + n] = v;
}

// ---------------------------------------------------------------------------
// Bilinear vert_align: feat (B,C,H,W) f32, verts (B,V,3) -> X (B*V, C) f16
// ---------------------------------------------------------------------------
__global__ void vert_align_kernel(const float* __restrict__ feat, const float* __restrict__ verts,
                                  _Float16* __restrict__ X, int C, int H, int W, long V) {
    long tid = (long)blockIdx.x * blockDim.x + threadIdx.x;
    const int nch = C >> 3;           // 8 channels per thread
    int  cc = (int)(tid % nch);
    long t  = tid / nch;
    long v  = t % V;
    int  b  = (int)(t / V);
    if (b >= BATCH) return;
    float vx = verts[((long)b * V + v) * 3 + 0];
    float vy = verts[((long)b * V + v) * 3 + 1];
    float px = (vx + 1.0f) * 0.5f * (float)(W - 1);
    float py = (vy + 1.0f) * 0.5f * (float)(H - 1);
    float x0 = floorf(px), y0 = floorf(py);
    float wx = px - x0,    wy = py - y0;
    int xi0 = min(max((int)x0, 0), W - 1);
    int xi1 = min(max((int)x0 + 1, 0), W - 1);
    int yi0 = min(max((int)y0, 0), H - 1);
    int yi1 = min(max((int)y0 + 1, 0), H - 1);
    float w00 = (1.f - wx) * (1.f - wy), w01 = wx * (1.f - wy);
    float w10 = (1.f - wx) * wy,         w11 = wx * wy;
    long outBase = ((long)b * V + v) * C + cc * 8;
    v8h out;
    #pragma unroll
    for (int j = 0; j < 8; ++j) {
        int c = cc * 8 + j;
        const float* f = feat + ((long)(b * C + c) * H) * W;
        float val = w00 * f[yi0 * W + xi0] + w01 * f[yi0 * W + xi1]
                  + w10 * f[yi1 * W + xi0] + w11 * f[yi1 * W + xi1];
        out[j] = (_Float16)val;
    }
    *(v8h*)(X + outBase) = out;
}

// ---------------------------------------------------------------------------
// Packed f16 atomic add (global_atomic_pk_add_f16)
// ---------------------------------------------------------------------------
__device__ __forceinline__ void pk_atomic_add(_Float16* p, v2h v) {
#if __has_builtin(__builtin_amdgcn_global_atomic_fadd_v2f16)
    (void)__builtin_amdgcn_global_atomic_fadd_v2f16((v2h*)p, v);
#else
    asm volatile("global_atomic_pk_add_f16 %0, %1, off"
                 :: "v"((v2h*)p), "v"(v) : "memory");
#endif
}

// Edge scatter-add in f16: neigh[b,e0,:] += x[b,e1,:]; neigh[b,e1,:] += x[b,e0,:]
// Thread = (edge, 8-channel chunk); loops over batches (L2-resident atomics).
__global__ void scatter_kernel(const _Float16* __restrict__ X, _Float16* __restrict__ NB,
                               const int* __restrict__ edges, int C, int E, long V) {
    long tid = (long)blockIdx.x * blockDim.x + threadIdx.x;
    const int chunks = C >> 3;
    long e = tid / chunks;
    int  c = (int)(tid % chunks) << 3;
    if (e >= E) return;
    int e0 = edges[2 * e + 0];
    int e1 = edges[2 * e + 1];
    #pragma unroll
    for (int b = 0; b < BATCH; ++b) {
        long r0 = ((long)b * V + e0) * C + c;
        long r1 = ((long)b * V + e1) * C + c;
        v8h d1 = *(const v8h*)(X + r1);
        v8h d0 = *(const v8h*)(X + r0);
        #pragma unroll
        for (int j = 0; j < 4; ++j) {
            v2h p1 = { d1[2 * j], d1[2 * j + 1] };
            v2h p0 = { d0[2 * j], d0[2 * j + 1] };
            pk_atomic_add(NB + r0 + 2 * j, p1);
            pk_atomic_add(NB + r1 + 2 * j, p0);
        }
    }
}

// ---------------------------------------------------------------------------
// Dual GEMM via WMMA: Y[M x Nout] = X @ W0 + NEI @ W1 + bias  (X/NEI f16)
// 8 waves/block, 16 rows per wave, all NT 16-col tiles per wave.
// Weights staged per 32-K chunk through LDS; A fragments prefetched one
// k-chunk ahead so global latency hides under the WMMA sequence.
// ---------------------------------------------------------------------------
__device__ __forceinline__ v16h load_a16(const _Float16* __restrict__ p) {
    v8h lo = *(const v8h*)(p);
    v8h hi = *(const v8h*)(p + 8);    // K run +8..+15
    return __builtin_shufflevector(lo, hi, 0, 1, 2, 3, 4, 5, 6, 7,
                                   8, 9, 10, 11, 12, 13, 14, 15);
}

template <int NT, bool HALF_OUT>
__global__ __launch_bounds__(256) void gc_gemm_kernel(
    const _Float16* __restrict__ X, const _Float16* __restrict__ NEI,
    const _Float16* __restrict__ W0f, const _Float16* __restrict__ W1f,
    const float* __restrict__ bias, void* __restrict__ Y,
    int K, int Nout) {
    __shared__ uint4 ldsw[2 * NT * 64];   // 2 matrices x NT tiles x 1KB
    const int tid  = threadIdx.x;
    const int lane = tid & 31;
    const int wave = tid >> 5;
    const int lr   = lane & 15;
    const int grp  = lane >> 4;
    const long m0  = ((long)blockIdx.x * 8 + wave) * 16;
    const int KT   = K >> 5;

    const _Float16* pX = X   + (m0 + lr) * (size_t)K + grp * 16;
    const _Float16* pN = NEI + (m0 + lr) * (size_t)K + grp * 16;

    v8f acc[NT];
    #pragma unroll
    for (int nt = 0; nt < NT; ++nt) {
        float bv = bias[nt * 16 + lr];
        v8f a;
        #pragma unroll
        for (int r = 0; r < 8; ++r) a[r] = bv;
        acc[nt] = a;
    }

    // prefetch A fragments for kt = 0
    v16h ax = load_a16(pX);
    v16h an = load_a16(pN);

    for (int kt = 0; kt < KT; ++kt) {
        __syncthreads();
        const uint4* s0 = (const uint4*)W0f + (size_t)kt * NT * 64;
        const uint4* s1 = (const uint4*)W1f + (size_t)kt * NT * 64;
        for (int i = tid; i < NT * 64; i += 256) {
            ldsw[i]           = s0[i];
            ldsw[NT * 64 + i] = s1[i];
        }
        __syncthreads();

        // prefetch next k-chunk A fragments (uniform branch, no divergence)
        v16h axn = ax, ann = an;
        if (kt + 1 < KT) {
            axn = load_a16(pX + (kt + 1) * 32);
            ann = load_a16(pN + (kt + 1) * 32);
        }

        const v16h* lp = (const v16h*)ldsw;
        #pragma unroll
        for (int nt = 0; nt < NT; ++nt) {
            v16h b0 = lp[nt * 32 + lane];
            v16h b1 = lp[NT * 32 + nt * 32 + lane];
            acc[nt] = __builtin_amdgcn_wmma_f32_16x16x32_f16(
                false, ax, false, b0, (short)0, acc[nt], false, false);
            acc[nt] = __builtin_amdgcn_wmma_f32_16x16x32_f16(
                false, an, false, b1, (short)0, acc[nt], false, false);
        }
        ax = axn;
        an = ann;
    }

    // C/D layout: VGPR r -> lanes 0-15 (M=r, N=lane), lanes 16-31 (M=8+r, N=lane-16)
    #pragma unroll
    for (int nt = 0; nt < NT; ++nt) {
        int col = nt * 16 + lr;
        if (col < Nout) {
            #pragma unroll
            for (int r = 0; r < 8; ++r) {
                long row = m0 + r + 8 * grp;
                if (HALF_OUT)
                    ((_Float16*)Y)[row * (long)Nout + col] = (_Float16)acc[nt][r];
                else
                    ((float*)Y)[row * (long)Nout + col] = acc[nt][r];
            }
        }
    }
}

// ---------------------------------------------------------------------------
extern "C" void kernel_launch(void* const* d_in, const int* in_sizes, int n_in,
                              void* d_out, int out_size, void* d_ws, size_t ws_size,
                              hipStream_t stream) {
    (void)in_sizes; (void)n_in; (void)out_size; (void)ws_size;
    const long V = NVERT;
    const long M = BATCH * V;            // 160000 rows, divisible by 128
    const int  E = NEDGE;

    // Only stage 3 contributes to the output (stage 1/2 results are discarded).
    const float* conv256  = (const float*)d_in[2];
    const float* vertices = (const float*)d_in[4];
    const int*   edges    = (const int*)d_in[5];
    const float* fW0 = (const float*)d_in[30];
    const float* fb0 = (const float*)d_in[31];
    const float* fW1 = (const float*)d_in[32];
    const float* fb1 = (const float*)d_in[33];
    const float* hW0 = (const float*)d_in[34];
    const float* hb0 = (const float*)d_in[35];
    const float* hW1 = (const float*)d_in[36];
    const float* hb1 = (const float*)d_in[37];
    const float* lW0 = (const float*)d_in[38];
    const float* lb0 = (const float*)d_in[39];
    const float* lW1 = (const float*)d_in[40];
    const float* lb1 = (const float*)d_in[41];

    char* ws = (char*)d_ws;
    size_t off = 0;
    auto alloc = [&](size_t bytes) -> void* {
        void* p = ws + off;
        off = (off + bytes + 255) & ~(size_t)255;
        return p;
    };
    _Float16* xBuf = (_Float16*)alloc(M * 256 * sizeof(_Float16)); // vert_align out
    _Float16* nBuf = (_Float16*)alloc(M * 256 * sizeof(_Float16)); // neigh scratch
    _Float16* hA   = (_Float16*)alloc(M * 192 * sizeof(_Float16));
    _Float16* hB   = (_Float16*)alloc(M * 192 * sizeof(_Float16));
    _Float16* fW0f = (_Float16*)alloc((size_t)8 * 12 * 512 * 2);
    _Float16* fW1f = (_Float16*)alloc((size_t)8 * 12 * 512 * 2);
    _Float16* hW0f = (_Float16*)alloc((size_t)12 * 6 * 12 * 512 * 2);
    _Float16* hW1f = (_Float16*)alloc((size_t)12 * 6 * 12 * 512 * 2);
    _Float16* lW0f = (_Float16*)alloc((size_t)6 * 1 * 512 * 2);
    _Float16* lW1f = (_Float16*)alloc((size_t)6 * 1 * 512 * 2);
    float* biasF = (float*)alloc(192 * sizeof(float));
    float* biasH = (float*)alloc(12 * 192 * sizeof(float));
    float* biasL = (float*)alloc(16 * sizeof(float));

    const long hFragStride = (long)6 * 12 * 512;   // f16 elements per hidden layer

    // ---- weight / bias prep ----
    swizzle_w_kernel<<<dim3(192, 1), 256, 0, stream>>>(fW0, fW0f, 256, 192, 12, 0, 0);
    swizzle_w_kernel<<<dim3(192, 1), 256, 0, stream>>>(fW1, fW1f, 256, 192, 12, 0, 0);
    swizzle_w_kernel<<<dim3(144, 12), 256, 0, stream>>>(hW0, hW0f, 192, 192, 12,
                                                        (long)192 * 192, hFragStride);
    swizzle_w_kernel<<<dim3(144, 12), 256, 0, stream>>>(hW1, hW1f, 192, 192, 12,
                                                        (long)192 * 192, hFragStride);
    swizzle_w_kernel<<<dim3(12, 1), 256, 0, stream>>>(lW0, lW0f, 192, 3, 1, 0, 0);
    swizzle_w_kernel<<<dim3(12, 1), 256, 0, stream>>>(lW1, lW1f, 192, 3, 1, 0, 0);
    bias_kernel<<<dim3(1, 1), 256, 0, stream>>>(fb0, fb1, biasF, 192, 192);
    bias_kernel<<<dim3(1, 12), 256, 0, stream>>>(hb0, hb1, biasH, 192, 192);
    bias_kernel<<<dim3(1, 1), 256, 0, stream>>>(lb0, lb1, biasL, 3, 16);

    // ---- vert_align(conv256) -> xBuf ----
    {
        long total = M * (256 / 8);
        vert_align_kernel<<<(int)((total + 255) / 256), 256, 0, stream>>>(
            conv256, vertices, xBuf, 256, 14, 14, V);
    }

    // ---- first GC: (M,256) -> (M,192) ----
    hipMemsetAsync(nBuf, 0, (size_t)M * 256 * sizeof(_Float16), stream);
    {
        long total = (long)E * (256 / 8);
        scatter_kernel<<<(int)((total + 255) / 256), 256, 0, stream>>>(
            xBuf, nBuf, edges, 256, E, V);
    }
    gc_gemm_kernel<12, true><<<(int)(M / 128), 256, 0, stream>>>(
        xBuf, nBuf, fW0f, fW1f, biasF, hA, 256, 192);

    // ---- 12 hidden GCs: (M,192) -> (M,192) ----
    _Float16* cur = hA;
    _Float16* nxt = hB;
    for (int l = 0; l < 12; ++l) {
        hipMemsetAsync(nBuf, 0, (size_t)M * 192 * sizeof(_Float16), stream);
        long total = (long)E * (192 / 8);
        scatter_kernel<<<(int)((total + 255) / 256), 256, 0, stream>>>(
            cur, nBuf, edges, 192, E, V);
        gc_gemm_kernel<12, true><<<(int)(M / 128), 256, 0, stream>>>(
            cur, nBuf, hW0f + (size_t)l * hFragStride, hW1f + (size_t)l * hFragStride,
            biasH + (size_t)l * 192, nxt, 192, 192);
        _Float16* t = cur; cur = nxt; nxt = t;
    }

    // ---- last GC: (M,192) -> (M,3) f32 into d_out ----
    hipMemsetAsync(nBuf, 0, (size_t)M * 192 * sizeof(_Float16), stream);
    {
        long total = (long)E * (192 / 8);
        scatter_kernel<<<(int)((total + 255) / 256), 256, 0, stream>>>(
            cur, nBuf, edges, 192, E, V);
    }
    gc_gemm_kernel<1, false><<<(int)(M / 128), 256, 0, stream>>>(
        cur, nBuf, lW0f, lW1f, biasL, d_out, 192, 3);
}
